// IGMC_15513421873718
// MI455X (gfx1250) — matrix operand order repaired and checked by reading, not compiled, not run on hardware
//
#include <hip/hip_runtime.h>
#include <hip/hip_bf16.h>

typedef __bf16 bf16_t;
typedef __attribute__((ext_vector_type(16))) __bf16 v16bf;
typedef __attribute__((ext_vector_type(8)))  __bf16 v8bf;
typedef __attribute__((ext_vector_type(8)))  float  v8f;

#define NNODES 100000
#define NEDGES 1600000
#define BPAIRS 2048
#define HDIM   32
#define NREL   5
#define NBASE  2
#define NLAYER 4

// ---------------------------------------------------------------------------
// small helpers
// ---------------------------------------------------------------------------
__global__ void zero_f32(float* __restrict__ p, int n) {
    int i = blockIdx.x * blockDim.x + threadIdx.x;
    if (i < n) p[i] = 0.0f;
}

// Fused bf16 weight matrix per layer l in 1..3: wcat[l][k][c], c<64 -> Vs, c>=64 -> loopW.
__global__ void prep_wcat(const float* __restrict__ Vs, const float* __restrict__ loops,
                          bf16_t* __restrict__ wcat) {
    int i = blockIdx.x * blockDim.x + threadIdx.x;  // 3*32*96
    if (i >= 3 * 32 * 96) return;
    int l = i / (32 * 96);
    int r = i % (32 * 96);
    int k = r / 96;
    int c = r % 96;
    float v;
    if (c < 64) {
        int b = c >> 5, o = c & 31;
        v = Vs[((size_t)(l * NBASE + b) * HDIM + k) * HDIM + o];
    } else {
        v = loops[((size_t)l * HDIM + k) * HDIM + (c - 64)];
    }
    wcat[i] = (bf16_t)v;
}

// w1 is [128(out),256(in)]; build B-matrix w1bf[k*128+n] = w1[n*256+k] in bf16.
__global__ void prep_w1(const float* __restrict__ w1, bf16_t* __restrict__ w1bf) {
    int i = blockIdx.x * blockDim.x + threadIdx.x;  // 256*128
    if (i >= 256 * 128) return;
    int k = i >> 7, n = i & 127;
    w1bf[i] = (bf16_t)w1[n * 256 + k];
}

// ---------------------------------------------------------------------------
// Layer 0: one-hot input (K=4), trivial scalar transform.
// xb[n,b,o] = sum_i x[n,i]*V0[b,i,o]; selfp[n,o] = sum_i x[n,i]*loop0[i,o]
// ---------------------------------------------------------------------------
__global__ void layer0_transform(const float* __restrict__ x, const float* __restrict__ V0,
                                 const float* __restrict__ loop0,
                                 float* __restrict__ xb, float* __restrict__ selfp) {
    int gid = blockIdx.x * blockDim.x + threadIdx.x;
    if (gid >= NNODES * HDIM) return;
    int n = gid >> 5, o = gid & 31;
    float x0 = x[n * 4 + 0], x1 = x[n * 4 + 1], x2 = x[n * 4 + 2], x3 = x[n * 4 + 3];
    float a0 = x0 * V0[(0 * 4 + 0) * 32 + o] + x1 * V0[(0 * 4 + 1) * 32 + o] +
               x2 * V0[(0 * 4 + 2) * 32 + o] + x3 * V0[(0 * 4 + 3) * 32 + o];
    float a1 = x0 * V0[(1 * 4 + 0) * 32 + o] + x1 * V0[(1 * 4 + 1) * 32 + o] +
               x2 * V0[(1 * 4 + 2) * 32 + o] + x3 * V0[(1 * 4 + 3) * 32 + o];
    float sp = x0 * loop0[0 * 32 + o] + x1 * loop0[1 * 32 + o] +
               x2 * loop0[2 * 32 + o] + x3 * loop0[3 * 32 + o];
    xb[(size_t)n * 64 + o]      = a0;
    xb[(size_t)n * 64 + 32 + o] = a1;
    selfp[(size_t)n * 32 + o]   = sp;
}

// ---------------------------------------------------------------------------
// Layers 1..3 node transform: [16x32]bf16 @ [32x96]bf16 -> f32 via v_wmma.
// One wave per 16-node tile; A reused across 6 N-tiles.
// A layout: lane = M(0-15 twice); elems0-7 = K{0-7|8-15}, elems8-15 = K{16-23|24-31}.
// B layout: lane = N col (0-15 twice); elems0-7 = K{0-7|8-15}, elems8-15 = K{16-23|24-31}.
// D layout: elem r -> M = r + 8*(lane>=16); N = lane&15.
// ---------------------------------------------------------------------------
__global__ __launch_bounds__(32)
void wmma_node_transform(const bf16_t* __restrict__ hbf, const bf16_t* __restrict__ wcat,
                         float* __restrict__ xb, float* __restrict__ selfp) {
    int tile = blockIdx.x;           // NNODES/16 tiles
    int lane = threadIdx.x;          // 0..31
    int m     = lane & 15;
    int khalf = (lane >> 4) * 8;
    const bf16_t* arow = hbf + (size_t)(tile * 16 + m) * HDIM;
    v8bf alo = *(const v8bf*)(arow + khalf);
    v8bf ahi = *(const v8bf*)(arow + 16 + khalf);
    v16bf a = __builtin_shufflevector(alo, ahi, 0, 1, 2, 3, 4, 5, 6, 7,
                                      8, 9, 10, 11, 12, 13, 14, 15);
    const bf16_t* brow = wcat + m;  // column N = lane&15
    int mrow = tile * 16 + ((lane >> 4) << 3);
#pragma unroll
    for (int nt = 0; nt < 6; ++nt) {
        v16bf b;
#pragma unroll
        for (int s = 0; s < 8; ++s) {
            b[s]     = brow[(size_t)(khalf + s) * 96 + nt * 16];
            b[s + 8] = brow[(size_t)(16 + khalf + s) * 96 + nt * 16];
        }
        v8f c = {};
        c = __builtin_amdgcn_wmma_f32_16x16x32_bf16(false, a, false, b,
                                                    (short)0, c, false, false);
        int col = nt * 16 + m;
#pragma unroll
        for (int r = 0; r < 8; ++r) {
            int row = mrow + r;
            if (col < 64) xb[(size_t)row * 64 + col]           = c[r];
            else          selfp[(size_t)row * 32 + (col - 64)] = c[r];
        }
    }
}

// ---------------------------------------------------------------------------
// Edge pass: 8 threads per edge, each covers 4 of 32 outputs.
// xb (25.6MB) and agg (12.8MB) are L2-resident on MI455X (192MB L2).
// ---------------------------------------------------------------------------
__global__ void edge_pass(const int* __restrict__ src, const int* __restrict__ dst,
                          const int* __restrict__ etype, const float* __restrict__ comp,
                          const float* __restrict__ xb, float* __restrict__ agg) {
    int gid = blockIdx.x * blockDim.x + threadIdx.x;
    int e = gid >> 3;
    if (e >= NEDGES) return;
    int part = gid & 7;
    int s = src[e], d = dst[e], t = etype[e];
    float c0 = comp[t * NBASE + 0];
    float c1 = comp[t * NBASE + 1];
    const float4* xr = (const float4*)(xb + (size_t)s * 64);
    float4 u = xr[part];
    float4 v = xr[8 + part];
    float4 mv;
    mv.x = c0 * u.x + c1 * v.x;
    mv.y = c0 * u.y + c1 * v.y;
    mv.z = c0 * u.z + c1 * v.z;
    mv.w = c0 * u.w + c1 * v.w;
    float* ap = agg + (size_t)d * 32 + part * 4;
    atomicAdd(ap + 0, mv.x);
    atomicAdd(ap + 1, mv.y);
    atomicAdd(ap + 2, mv.z);
    atomicAdd(ap + 3, mv.w);
}

// h = tanh(agg + selfp + bias); keep f32 in states, bf16 copy for next WMMA A.
__global__ void finalize_layer(const float* __restrict__ agg, const float* __restrict__ selfp,
                               const float* __restrict__ bias, float* __restrict__ states_l,
                               bf16_t* __restrict__ hbf) {
    int gid = blockIdx.x * blockDim.x + threadIdx.x;
    if (gid >= NNODES * HDIM) return;
    int o = gid & 31;
    float t = tanhf(agg[gid] + selfp[gid] + bias[o]);
    states_l[gid] = t;
    hbf[gid] = (bf16_t)t;
}

// feat[i, j] = j<128 ? states[j/32][i][j%32] : states[(j-128)/32][B+i][(j-128)%32]
__global__ void feat_prep(const float* __restrict__ states, bf16_t* __restrict__ featbf) {
    int gid = blockIdx.x * blockDim.x + threadIdx.x;
    if (gid >= BPAIRS * 256) return;
    int i = gid >> 8, j = gid & 255;
    int node = (j < 128) ? i : (BPAIRS + i);
    int jj   = (j < 128) ? j : (j - 128);
    int l = jj >> 5, o = jj & 31;
    featbf[gid] = (bf16_t)states[(size_t)l * NNODES * HDIM + (size_t)node * HDIM + o];
}

// hid = relu(feat @ w1.T + b1): [2048x256]@[256x128], K=256 in 8 WMMA steps.
__global__ __launch_bounds__(32)
void mlp1_wmma(const bf16_t* __restrict__ featbf, const bf16_t* __restrict__ w1bf,
               const float* __restrict__ b1, float* __restrict__ hid) {
    int blk = blockIdx.x;            // 128 mtiles * 8 ntiles
    int mt = blk >> 3, nt = blk & 7;
    int lane = threadIdx.x;
    int m     = lane & 15;
    int khalf = (lane >> 4) * 8;
    const bf16_t* arow = featbf + (size_t)(mt * 16 + m) * 256;
    v8f c = {};
#pragma unroll
    for (int kk = 0; kk < 8; ++kk) {
        v8bf alo = *(const v8bf*)(arow + kk * 32 + khalf);
        v8bf ahi = *(const v8bf*)(arow + kk * 32 + 16 + khalf);
        v16bf a = __builtin_shufflevector(alo, ahi, 0, 1, 2, 3, 4, 5, 6, 7,
                                          8, 9, 10, 11, 12, 13, 14, 15);
        v16bf b;
#pragma unroll
        for (int s = 0; s < 8; ++s) {
            b[s]     = w1bf[(size_t)(kk * 32 + khalf + s) * 128 + nt * 16 + m];
            b[s + 8] = w1bf[(size_t)(kk * 32 + 16 + khalf + s) * 128 + nt * 16 + m];
        }
        c = __builtin_amdgcn_wmma_f32_16x16x32_bf16(false, a, false, b,
                                                    (short)0, c, false, false);
    }
    int col = nt * 16 + m;
    float bias = b1[col];
    int mrow = mt * 16 + ((lane >> 4) << 3);
#pragma unroll
    for (int r = 0; r < 8; ++r) {
        float v = c[r] + bias;
        hid[(size_t)(mrow + r) * 128 + col] = v > 0.0f ? v : 0.0f;
    }
}

__global__ void mlp2(const float* __restrict__ hid, const float* __restrict__ w2,
                     const float* __restrict__ b2, float* __restrict__ out) {
    int i = blockIdx.x * blockDim.x + threadIdx.x;
    if (i >= BPAIRS) return;
    float s = b2[0];
#pragma unroll 8
    for (int n = 0; n < 128; ++n) s += hid[(size_t)i * 128 + n] * w2[n];
    out[i] = s;
}

// ---------------------------------------------------------------------------
extern "C" void kernel_launch(void* const* d_in, const int* in_sizes, int n_in,
                              void* d_out, int out_size, void* d_ws, size_t ws_size,
                              hipStream_t stream) {
    (void)in_sizes; (void)n_in; (void)out_size; (void)ws_size;
    const float* x      = (const float*)d_in[0];
    const int*   src    = (const int*)d_in[1];
    const int*   dst    = (const int*)d_in[2];
    const int*   etype  = (const int*)d_in[3];
    const float* V0     = (const float*)d_in[4];
    const float* comp0  = (const float*)d_in[5];
    const float* loop0  = (const float*)d_in[6];
    const float* bias0  = (const float*)d_in[7];
    const float* Vs     = (const float*)d_in[8];
    const float* comps  = (const float*)d_in[9];
    const float* loops  = (const float*)d_in[10];
    const float* biases = (const float*)d_in[11];
    const float* w1     = (const float*)d_in[12];
    const float* b1     = (const float*)d_in[13];
    const float* w2     = (const float*)d_in[14];
    const float* b2     = (const float*)d_in[15];
    float* out = (float*)d_out;

    char* ws = (char*)d_ws;
    size_t off = 0;
    auto carve = [&](size_t bytes) -> char* {
        char* p = ws + off;
        off += (bytes + 255) & ~(size_t)255;
        return p;
    };
    float*  xb     = (float*) carve((size_t)NNODES * 64 * 4);
    float*  selfp  = (float*) carve((size_t)NNODES * 32 * 4);
    float*  agg    = (float*) carve((size_t)NNODES * 32 * 4);
    float*  states = (float*) carve((size_t)NLAYER * NNODES * 32 * 4);
    bf16_t* hbf    = (bf16_t*)carve((size_t)NNODES * 32 * 2);
    bf16_t* wcat   = (bf16_t*)carve((size_t)3 * 32 * 96 * 2);
    bf16_t* featbf = (bf16_t*)carve((size_t)BPAIRS * 256 * 2);
    bf16_t* w1bf   = (bf16_t*)carve((size_t)256 * 128 * 2);
    float*  hid    = (float*) carve((size_t)BPAIRS * 128 * 4);

    prep_wcat<<<(3 * 32 * 96 + 255) / 256, 256, 0, stream>>>(Vs, loops, wcat);
    prep_w1<<<(256 * 128 + 255) / 256, 256, 0, stream>>>(w1, w1bf);

    const int nNH = NNODES * HDIM;
    for (int l = 0; l < NLAYER; ++l) {
        if (l == 0)
            layer0_transform<<<(nNH + 255) / 256, 256, 0, stream>>>(x, V0, loop0, xb, selfp);
        else
            wmma_node_transform<<<NNODES / 16, 32, 0, stream>>>(
                hbf, wcat + (size_t)(l - 1) * 32 * 96, xb, selfp);
        zero_f32<<<(nNH + 255) / 256, 256, 0, stream>>>(agg, nNH);
        const float* comp = (l == 0) ? comp0 : (comps + (size_t)(l - 1) * NREL * NBASE);
        edge_pass<<<(NEDGES * 8 + 255) / 256, 256, 0, stream>>>(src, dst, etype, comp, xb, agg);
        const float* bias = (l == 0) ? bias0 : (biases + (size_t)(l - 1) * HDIM);
        finalize_layer<<<(nNH + 255) / 256, 256, 0, stream>>>(
            agg, selfp, bias, states + (size_t)l * nNH, hbf);
    }
    feat_prep<<<(BPAIRS * 256 + 255) / 256, 256, 0, stream>>>(states, featbf);
    mlp1_wmma<<<128 * 8, 32, 0, stream>>>(featbf, w1bf, b1, hid);
    mlp2<<<(BPAIRS + 255) / 256, 256, 0, stream>>>(hid, w2, b2, out);
}
